// AudioXMMDiTSelfAttention_27212912787911
// MI455X (gfx1250) — compile-verified, matrix operand order checked by reading, not run on hardware
//
#include <hip/hip_runtime.h>
#include <hip/hip_bf16.h>
#include <math.h>

#define DIMV   1536
#define NH     24
#define HD     64
#define S_LEN  4096
#define B_SZ   4
#define N3     (3*DIMV)

typedef __attribute__((ext_vector_type(16))) __bf16 v16bf;
typedef __attribute__((ext_vector_type(8)))  __bf16 v8bf;
typedef __attribute__((ext_vector_type(8)))  float  v8f;

// ---- CDNA5 async global->LDS copy (ASYNCcnt-tracked DMA, no VGPR bounce) ----
__device__ __forceinline__ unsigned lds_off_u32(const void* p) {
    // generic LDS address = {shared_aperture_hi32, lds_offset32}
    return (unsigned)(uintptr_t)p;
}
__device__ __forceinline__ void async_ld_b128(const void* gptr, unsigned lds_off) {
    asm volatile("global_load_async_to_lds_b128 %0, %1, off"
                 :: "v"(lds_off), "v"((unsigned long long)(uintptr_t)gptr)
                 : "memory");
}
#define WAIT_ASYNC(n) asm volatile("s_wait_asynccnt " #n ::: "memory")

// Load a 16-element bf16 WMMA operand fragment (A or B role) from LDS.
// ISA 16-bit 16x32 layout: lanes 0-15 hold k {0..7, 16..23},
// lanes 16-31 hold k {8..15, 24..31}; koff = (lane<16)?0:8.
__device__ __forceinline__ v16bf load_frag(const __bf16* base, int koff) {
    v8bf lo = *(const v8bf*)(base + koff);
    v8bf hi = *(const v8bf*)(base + koff + 16);
    v16bf r;
#pragma unroll
    for (int i = 0; i < 8; ++i) { r[i] = lo[i]; r[8 + i] = hi[i]; }
    return r;
}

__device__ __forceinline__ v8f wmma_bf16(v16bf a, v16bf b, v8f c) {
    return __builtin_amdgcn_wmma_f32_16x16x32_bf16(
        false, a, false, b, (short)0, c, false, false);
}

// ---------------------------------------------------------------- convert x
__global__ void cvt_f32_bf16(const float* __restrict__ src,
                             __bf16* __restrict__ dst, int n) {
    int i = blockIdx.x * blockDim.x + threadIdx.x;
    int stride = gridDim.x * blockDim.x;
    for (; i < n; i += stride) dst[i] = (__bf16)src[i];
}

// ------------------------------------------------- one-time W transpose (f32->bf16)
// wt[n][k] = w[k][n];   32x32 tiles through LDS, both sides coalesced.
__global__ __launch_bounds__(256)
void transpose_w_bf16(const float* __restrict__ w, __bf16* __restrict__ wt) {
    __shared__ float tile[32][33];
    const int k0 = blockIdx.x * 32;            // over DIMV
    const int n0 = blockIdx.y * 32;            // over N3
    const int tx = threadIdx.x & 31, ty = threadIdx.x >> 5;   // 32x8
#pragma unroll
    for (int i = 0; i < 4; ++i) {
        int k = ty + i * 8;
        tile[k][tx] = w[(size_t)(k0 + k) * N3 + n0 + tx];
    }
    __syncthreads();
#pragma unroll
    for (int i = 0; i < 4; ++i) {
        int n = ty + i * 8;
        wt[(size_t)(n0 + n) * DIMV + k0 + tx] = (__bf16)tile[tx][n];
    }
}

// ---------------------------------------------------------------- QKV GEMM
// 128x64 C tile (N-tile == one head). 4 waves, 32x64 per wave (8 WMMA per
// 6 fragment loads). Double-buffered async tile staging.
__global__ __launch_bounds__(128)
void qkv_gemm(const __bf16* __restrict__ xb, const __bf16* __restrict__ wt,
              const float* __restrict__ bias, const float* __restrict__ cosT,
              const float* __restrict__ sinT,
              __bf16* __restrict__ qh, __bf16* __restrict__ kh,
              __bf16* __restrict__ vt) {
    __shared__ __attribute__((aligned(16))) __bf16 As[2][128 * 32];  // [row][k]
    __shared__ __attribute__((aligned(16))) __bf16 Bs[2][64 * 32];   // [n][k]
    __shared__ __attribute__((aligned(16))) float  Ct[128 * 64];

    const int tid  = threadIdx.x;
    const int lane = tid & 31;
    const int wv   = tid >> 5;                 // 4 waves
    const int m0   = blockIdx.y * 128;
    const int n0   = blockIdx.x * 64;
    const int koff = (lane < 16) ? 0 : 8;
    const int ln   = lane & 15;
    const int hi   = lane >> 4;
    const int rbase = wv * 32;                 // wave's 32-row strip

    // async staging: 4 A-chunks + 2 B-chunks per thread per tile (6 issues)
    auto issue_tile = [&](int buf, int k0) {
#pragma unroll
        for (int c = tid; c < 512; c += 128) {
            int r = c >> 2, kc = (c & 3) * 8;
            async_ld_b128(&xb[(size_t)(m0 + r) * DIMV + k0 + kc],
                          lds_off_u32(&As[buf][r * 32 + kc]));
        }
#pragma unroll
        for (int c = tid; c < 256; c += 128) {
            int n = c >> 2, kc = (c & 3) * 8;
            async_ld_b128(&wt[(size_t)(n0 + n) * DIMV + k0 + kc],
                          lds_off_u32(&Bs[buf][n * 32 + kc]));
        }
    };

    v8f acc[2][4];
#pragma unroll
    for (int rb = 0; rb < 2; ++rb)
#pragma unroll
        for (int j = 0; j < 4; ++j) acc[rb][j] = v8f{};

    issue_tile(0, 0);
    const int NK = DIMV / 32;                  // 48
    for (int kt = 0; kt < NK; ++kt) {
        const int cur = kt & 1;
        if (kt + 1 < NK) { issue_tile(cur ^ 1, (kt + 1) * 32); WAIT_ASYNC(0x6); }
        else             { WAIT_ASYNC(0x0); }
        __syncthreads();                       // all waves' tile chunks visible

        v16bf a0 = load_frag(&As[cur][(rbase + ln) * 32], koff);
        v16bf a1 = load_frag(&As[cur][(rbase + 16 + ln) * 32], koff);
#pragma unroll
        for (int j = 0; j < 4; ++j) {
            v16bf b = load_frag(&Bs[cur][(j * 16 + ln) * 32], koff);
            acc[0][j] = wmma_bf16(a0, b, acc[0][j]);
            acc[1][j] = wmma_bf16(a1, b, acc[1][j]);
        }
        __syncthreads();                       // reads done before buf reuse
    }

    // spill C tile for row-wise epilogue
#pragma unroll
    for (int rb = 0; rb < 2; ++rb)
#pragma unroll
        for (int j = 0; j < 4; ++j)
#pragma unroll
            for (int e = 0; e < 8; ++e)
                Ct[(rbase + rb * 16 + e + hi * 8) * 64 + j * 16 + ln] =
                    acc[rb][j][e];
    __syncthreads();

    const int region = n0 / DIMV;              // 0=q 1=k 2=v (tile never straddles)
    const int h      = (n0 - region * DIMV) >> 6;
    const int bb     = m0 >> 12;               // 128-row tile stays in one batch
    const int s0     = m0 & 4095;

    if (region == 2) {
        // V: store transposed [B,H,D,S]; each thread: one (d, 64-s) strip
        const int d  = tid >> 1;
        const int sc = (tid & 1) * 64;
        const float bv = bias[n0 + d];
        const size_t vbase = ((size_t)(bb * NH + h) * HD + d) * S_LEN + s0 + sc;
#pragma unroll
        for (int i = 0; i < 64; i += 8) {
            v8bf p;
#pragma unroll
            for (int j = 0; j < 8; ++j)
                p[j] = (__bf16)(Ct[(sc + i + j) * 64 + d] + bv);
            *(v8bf*)&vt[vbase + i] = p;
        }
    } else {
        // Q/K: bias + RMSNorm + interleaved RoPE, one thread per row (128)
        const int row = tid;
        const int s   = s0 + row;
        float y[64];
#pragma unroll
        for (int c = 0; c < 64; ++c) y[c] = Ct[row * 64 + c] + bias[n0 + c];
        float ms = 0.f;
#pragma unroll
        for (int c = 0; c < 64; ++c) ms += y[c] * y[c];
        float rr = rsqrtf(ms * (1.0f / 64.0f) + 1e-6f);
#pragma unroll
        for (int c = 0; c < 64; ++c) y[c] *= rr;
#pragma unroll
        for (int i = 0; i < 32; ++i) {
            float cv = cosT[s * HD + 2 * i];
            float sv = sinT[s * HD + 2 * i];
            float x1 = y[2 * i], x2 = y[2 * i + 1];
            y[2 * i]     = x1 * cv - x2 * sv;
            y[2 * i + 1] = x2 * cv + x1 * sv;
        }
        __bf16* dst = (region == 0) ? qh : kh;
        size_t base = ((size_t)(bb * NH + h) * S_LEN + s) * HD;
#pragma unroll
        for (int c = 0; c < 64; ++c) dst[base + c] = (__bf16)y[c];
    }
}

// ---------------------------------------------------------------- attention
// Flash-style, 64 q-rows/block, 64-key tiles, double-buffered async K/V.
__global__ __launch_bounds__(256)
void flash_attn(const __bf16* __restrict__ qh, const __bf16* __restrict__ kh,
                const __bf16* __restrict__ vt, float* __restrict__ out) {
    __shared__ __attribute__((aligned(16))) __bf16 Ks[2][64 * 64]; // [key][d]
    __shared__ __attribute__((aligned(16))) __bf16 Vs[2][64 * 64]; // [d][key]
    __shared__ __attribute__((aligned(16))) __bf16 Ps[64 * 64];    // [q][key]
    __shared__ __attribute__((aligned(16))) float  Ss[64 * 64];    // scores
    __shared__ float alpha_s[64];
    __shared__ float linv_s[64];
    __bf16* Qs = (__bf16*)Ss;   // Q tile overlays score buffer (barrier-separated)

    const int tid  = threadIdx.x;
    const int lane = tid & 31;
    const int wv   = tid >> 5;
    const int bh   = blockIdx.y;
    const int bb   = bh / NH, h = bh % NH;
    const int q0   = blockIdx.x * 64;
    const int koff = (lane < 16) ? 0 : 8;
    const int ln   = lane & 15;
    const int hi   = lane >> 4;
    const int rm   = wv >> 1;
    const int ch   = wv & 1;
    const int rbase = rm * 16;
    const size_t head_base = (size_t)bh * S_LEN * HD;   // qh/kh [B,H,S,D]
    const size_t vhd_base  = (size_t)bh * HD * S_LEN;   // vt    [B,H,D,S]
    const float scale = 0.125f;

    // per-thread async chunk coords (2 K + 2 V issues per tile)
    const int r0 = tid >> 3,         dc0 = (tid & 7) * 8;
    const int r1 = (tid + 256) >> 3, dc1 = dc0;

    auto issue_kv = [&](int buf, int k0) {
        async_ld_b128(&kh[head_base + (size_t)(k0 + r0) * HD + dc0],
                      lds_off_u32(&Ks[buf][r0 * 64 + dc0]));
        async_ld_b128(&kh[head_base + (size_t)(k0 + r1) * HD + dc1],
                      lds_off_u32(&Ks[buf][r1 * 64 + dc1]));
        async_ld_b128(&vt[vhd_base + (size_t)r0 * S_LEN + k0 + dc0],
                      lds_off_u32(&Vs[buf][r0 * 64 + dc0]));
        async_ld_b128(&vt[vhd_base + (size_t)r1 * S_LEN + k0 + dc1],
                      lds_off_u32(&Vs[buf][r1 * 64 + dc1]));
    };

    // Q tile (async; loads complete in issue order -> wait<=4 covers Q)
    async_ld_b128(&qh[head_base + (size_t)(q0 + r0) * HD + dc0],
                  lds_off_u32(&Qs[r0 * 64 + dc0]));
    async_ld_b128(&qh[head_base + (size_t)(q0 + r1) * HD + dc1],
                  lds_off_u32(&Qs[r1 * 64 + dc1]));
    issue_kv(0, 0);
    WAIT_ASYNC(0x4);
    __syncthreads();
    v16bf qa0 = load_frag(&Qs[(rbase + ln) * 64], koff);
    v16bf qa1 = load_frag(&Qs[(rbase + ln) * 64 + 32], koff);

    v8f o[2]; o[0] = v8f{}; o[1] = v8f{};
    float m_r = -INFINITY, l_r = 0.f;       // live in threads 0..63

    const int NKT = S_LEN / 64;             // 64
    for (int kt = 0; kt < NKT; ++kt) {
        const int cur = kt & 1;
        if (kt + 1 < NKT) { issue_kv(cur ^ 1, (kt + 1) * 64); WAIT_ASYNC(0x4); }
        else              { WAIT_ASYNC(0x0); }
        __syncthreads();

        // S = Q K^T * scale
#pragma unroll
        for (int jj = 0; jj < 2; ++jj) {
            int nb = ch * 32 + jj * 16;
            v16bf b0 = load_frag(&Ks[cur][(nb + ln) * 64], koff);
            v16bf b1 = load_frag(&Ks[cur][(nb + ln) * 64 + 32], koff);
            v8f sacc = v8f{};
            sacc = wmma_bf16(qa0, b0, sacc);
            sacc = wmma_bf16(qa1, b1, sacc);
#pragma unroll
            for (int e = 0; e < 8; ++e)
                Ss[(rbase + e + hi * 8) * 64 + nb + ln] = sacc[e] * scale;
        }
        __syncthreads();

        // online softmax: one thread per query row
        if (tid < 64) {
            float mx = m_r;
#pragma unroll
            for (int j = 0; j < 64; ++j) mx = fmaxf(mx, Ss[tid * 64 + j]);
            float al = __expf(m_r - mx);
            float sum = 0.f;
#pragma unroll
            for (int j = 0; j < 64; ++j) {
                float p = __expf(Ss[tid * 64 + j] - mx);
                sum += p;
                Ps[tid * 64 + j] = (__bf16)p;
            }
            l_r = l_r * al + sum;
            m_r = mx;
            alpha_s[tid] = al;
        }
        __syncthreads();

        // O = O*alpha + P V
        v16bf pa0 = load_frag(&Ps[(rbase + ln) * 64], koff);
        v16bf pa1 = load_frag(&Ps[(rbase + ln) * 64 + 32], koff);
#pragma unroll
        for (int jj = 0; jj < 2; ++jj) {
            int nb = ch * 32 + jj * 16;
            v16bf b0 = load_frag(&Vs[cur][(nb + ln) * 64], koff);
            v16bf b1 = load_frag(&Vs[cur][(nb + ln) * 64 + 32], koff);
#pragma unroll
            for (int e = 0; e < 8; ++e) o[jj][e] *= alpha_s[rbase + e + hi * 8];
            o[jj] = wmma_bf16(pa0, b0, o[jj]);
            o[jj] = wmma_bf16(pa1, b1, o[jj]);
        }
        __syncthreads();
    }

    if (tid < 64) linv_s[tid] = 1.0f / l_r;
    __syncthreads();

#pragma unroll
    for (int jj = 0; jj < 2; ++jj) {
        int nb = ch * 32 + jj * 16;
#pragma unroll
        for (int e = 0; e < 8; ++e) {
            int row = rbase + e + hi * 8;
            out[((size_t)bb * S_LEN + q0 + row) * DIMV + h * HD + nb + ln] =
                o[jj][e] * linv_s[row];
        }
    }
}

// ---------------------------------------------------------------- launcher
extern "C" void kernel_launch(void* const* d_in, const int* in_sizes, int n_in,
                              void* d_out, int out_size, void* d_ws, size_t ws_size,
                              hipStream_t stream) {
    const float* x    = (const float*)d_in[0];
    const float* cosT = (const float*)d_in[1];
    const float* sinT = (const float*)d_in[2];
    const float* w    = (const float*)d_in[3];
    const float* bias = (const float*)d_in[4];
    float* out = (float*)d_out;

    const size_t n_x  = (size_t)B_SZ * S_LEN * DIMV;
    const size_t n_w  = (size_t)DIMV * N3;
    const size_t n_hd = (size_t)B_SZ * NH * S_LEN * HD;

    __bf16* xb = (__bf16*)d_ws;
    __bf16* wt = xb + n_x;
    __bf16* qh = wt + n_w;
    __bf16* kh = qh + n_hd;
    __bf16* vt = kh + n_hd;

    cvt_f32_bf16<<<2048, 256, 0, stream>>>(x, xb, (int)n_x);
    transpose_w_bf16<<<dim3(DIMV / 32, N3 / 32), 256, 0, stream>>>(w, wt);

    qkv_gemm<<<dim3(N3 / 64, (B_SZ * S_LEN) / 128), 128, 0, stream>>>(
        xb, wt, bias, cosT, sinT, qh, kh, vt);

    flash_attn<<<dim3(S_LEN / 64, B_SZ * NH), 256, 0, stream>>>(qh, kh, vt, out);
}
